// Informer_74380243632544
// MI455X (gfx1250) — matrix operand order repaired and checked by reading, not compiled, not run on hardware
//
#include <hip/hip_runtime.h>
#include <math.h>
#include <stdint.h>

// ---------------------------------------------------------------------------
// Informer forward for MI455X (gfx1250, wave32).
// All dense 512-wide GEMMs run on v_wmma_f32_16x16x32_f16 (f32 accumulate),
// with the A stripe staged per-block in LDS (f16) and read via ds_load_b128.
// Assumes the harness flattens setup_inputs() as a JAX pytree (sorted dict
// keys) -> 58 leaves; see the PARAM enum below for the exact mapping.
// NOTE: prob-sparse sampling uses a deterministic hash in place of JAX
// threefry (no runtime check available); everything else follows the ref.
// ---------------------------------------------------------------------------

#define DM   512
#define NH   8
#define HD   64
#define BATCH 4
#define LENC 2048
#define LDEC 1024
#define PRED 512

#define NEG_BIG (-1.0e30f)

typedef __attribute__((ext_vector_type(16))) _Float16 v16h;
typedef __attribute__((ext_vector_type(8)))  _Float16 v8h;
typedef __attribute__((ext_vector_type(8)))  float    v8f;

__device__ __forceinline__ uint32_t mixhash(uint32_t x) {
  x ^= x >> 16; x *= 0x7feb352dU;
  x ^= x >> 15; x *= 0x846ca68bU;
  x ^= x >> 16; return x;
}

// ---------------------------------------------------------------------------
// WMMA GEMM: Y[M,N] = X[M,K] * W[N,K]^T + bias[N], optional exact GELU.
// N must be 512 (true at every call site). Block = 8 waves; blockIdx = 16-row
// stripe (tM); wave w owns N-macro-tile w (64 cols, 4 accumulators).
// A stripe (16 x K) is converted to f16 once into LDS and shared by all 8
// waves; per K-step each lane reassembles its ISA 7.12.2 A fragment from two
// aligned 16B LDS chunks. B streams from global (L2-resident weights).
// Requires M%16==0, K%32==0.
// ---------------------------------------------------------------------------
__device__ __forceinline__ v16h load_b_frag(const float* __restrict__ wr, int k0, int hi) {
  v16h b;
#pragma unroll
  for (int j = 0; j < 8; ++j) {
    const int kb = k0 + 2 * j + hi * 8 + (j >= 4 ? 8 : 0);
    b[2 * j]     = (_Float16)wr[kb];
    b[2 * j + 1] = (_Float16)wr[kb + 1];
  }
  return b;
}

__global__ __launch_bounds__(256)
void gemm_wmma_kernel(const float* __restrict__ X, const float* __restrict__ W,
                      const float* __restrict__ bias, float* __restrict__ Y,
                      int M, int N, int K, int act) {
  extern __shared__ _Float16 As[];          // 16 * K f16 (16..48 KB)
  const int lane = threadIdx.x & 31;
  const int wave = threadIdx.x >> 5;        // == tN (N = 512 -> 8 macro-tiles)
  const int lo = lane & 15;
  const int hi = lane >> 4;
  const int tM = blockIdx.x;

  // Stage the 16 x K A stripe into LDS as f16 (coalesced, one barrier).
  for (int row = 0; row < 16; ++row) {
    const float* src = X + (size_t)(tM * 16 + row) * K;
    _Float16* dst = As + row * K;
    for (int k = threadIdx.x; k < K; k += 256) dst[k] = (_Float16)src[k];
  }
  __syncthreads();

  const float* w0 = W + (size_t)(wave * 64 + lo) * K;
  const _Float16* arow = As + lo * K;

  v8f acc0 = {}, acc1 = {}, acc2 = {}, acc3 = {};
  for (int k0 = 0; k0 < K; k0 += 32) {
    // A fragment: elements 0..7 <- K = k0+hi*8 .. +7 ; 8..15 <- +16.
    const v8h alo = *(const v8h*)(arow + k0 + hi * 8);
    const v8h ahi = *(const v8h*)(arow + k0 + 16 + hi * 8);
    const v16h a = __builtin_shufflevector(alo, ahi, 0, 1, 2, 3, 4, 5, 6, 7,
                                           8, 9, 10, 11, 12, 13, 14, 15);
    const v16h b0 = load_b_frag(w0 + (size_t)0  * K, k0, hi);
    const v16h b1 = load_b_frag(w0 + (size_t)16 * K, k0, hi);
    const v16h b2 = load_b_frag(w0 + (size_t)32 * K, k0, hi);
    const v16h b3 = load_b_frag(w0 + (size_t)48 * K, k0, hi);
    acc0 = __builtin_amdgcn_wmma_f32_16x16x32_f16(false, a, false, b0, (short)0, acc0, false, false);
    acc1 = __builtin_amdgcn_wmma_f32_16x16x32_f16(false, a, false, b1, (short)0, acc1, false, false);
    acc2 = __builtin_amdgcn_wmma_f32_16x16x32_f16(false, a, false, b2, (short)0, acc2, false, false);
    acc3 = __builtin_amdgcn_wmma_f32_16x16x32_f16(false, a, false, b3, (short)0, acc3, false, false);
  }

  const int colBase = wave * 64;
#pragma unroll
  for (int t = 0; t < 4; ++t) {
    const v8f accv = (t == 0) ? acc0 : (t == 1) ? acc1 : (t == 2) ? acc2 : acc3;
    const int col = colBase + t * 16 + lo;
    const float bv = bias ? bias[col] : 0.0f;
#pragma unroll
    for (int r = 0; r < 8; ++r) {
      const int orow = tM * 16 + r + (hi ? 8 : 0);
      float v = accv[r] + bv;
      if (act == 1) v = 0.5f * v * (1.0f + erff(v * 0.70710678118f));  // exact GELU
      Y[(size_t)orow * N + col] = v;
    }
  }
}

// ---------------------------------------------------------------------------
// Embedding: circular tok-conv over C_IN=7 + positional + 4 temporal sinusoids
// (all sinusoid tables share the same d=512 analytic formula).
// ---------------------------------------------------------------------------
__global__ void embed_kernel(const float* __restrict__ x, const int* __restrict__ mark,
                             const float* __restrict__ tokw, float* __restrict__ out,
                             int B, int L) {
  const size_t i = (size_t)blockIdx.x * blockDim.x + threadIdx.x;
  const size_t total = (size_t)B * L * DM;
  if (i >= total) return;
  const int o = (int)(i % DM);
  size_t r = i / DM;
  const int l = (int)(r % L);
  const int b = (int)(r / L);

  const int i2 = o & ~1;
  const float div = __expf(-(float)i2 * (9.210340371976184f / (float)DM));
  const bool even = !(o & 1);
  const int* mk = mark + ((size_t)b * L + l) * 4;

  float acc;
  {
    float ang = (float)l * div;
    acc = even ? __sinf(ang) : __cosf(ang);      // POS_T
  }
#pragma unroll
  for (int j = 0; j < 4; ++j) {
    float ang = (float)mk[j] * div;
    acc += even ? __sinf(ang) : __cosf(ang);     // MONTH/DAY/WEEK/HOUR
  }
  const float* wrow = tokw + (size_t)o * 21;     // (512,7,3)
#pragma unroll
  for (int c = 0; c < 7; ++c) {
#pragma unroll
    for (int k = 0; k < 3; ++k) {
      int src = l + k - 1;
      src = (src + L) % L;
      acc += x[((size_t)b * L + src) * 7 + c] * wrow[c * 3 + k];
    }
  }
  out[i] = acc;
}

// ---------------------------------------------------------------------------
// LayerNorm over last dim (512); optional fused residual add. Wave per row.
// ---------------------------------------------------------------------------
__global__ __launch_bounds__(256)
void layernorm_kernel(const float* __restrict__ a, const float* __restrict__ resid,
                      const float* __restrict__ g, const float* __restrict__ b,
                      float* __restrict__ out, int rows) {
  const int row = blockIdx.x * 8 + (threadIdx.x >> 5);
  if (row >= rows) return;
  const int lane = threadIdx.x & 31;
  const float* ar = a + (size_t)row * DM;
  const float* rr = resid ? resid + (size_t)row * DM : nullptr;
  float v[16];
  float s = 0.f;
#pragma unroll
  for (int i = 0; i < 16; ++i) {
    float x = ar[i * 32 + lane];
    if (rr) x += rr[i * 32 + lane];
    v[i] = x; s += x;
  }
#pragma unroll
  for (int o = 16; o > 0; o >>= 1) s += __shfl_xor(s, o, 32);
  const float mean = s * (1.0f / DM);
  float vs = 0.f;
#pragma unroll
  for (int i = 0; i < 16; ++i) { float d = v[i] - mean; vs += d * d; }
#pragma unroll
  for (int o = 16; o > 0; o >>= 1) vs += __shfl_xor(vs, o, 32);
  const float inv = rsqrtf(vs * (1.0f / DM) + 1e-5f);
  float* orow = out + (size_t)row * DM;
#pragma unroll
  for (int i = 0; i < 16; ++i) {
    const int c = i * 32 + lane;
    orow[c] = (v[i] - mean) * inv * g[c] + b[c];
  }
}

// ---------------------------------------------------------------------------
// Prob-sparse: sparsity measure M[b,h,l] = max_s(q.k_samp) - sum_s/Lk.
// One wave per (b,h,l); hashed sample indices shared across (b,h).
// q,k layout: (B,L,H,HD) i.e. GEMM output rows.
// ---------------------------------------------------------------------------
__global__ __launch_bounds__(256)
void prob_meas_kernel(const float* __restrict__ q, const float* __restrict__ k,
                      float* __restrict__ Mm, int B, int H, int Lq, int Lk,
                      int U, uint32_t seed) {
  const int gw = blockIdx.x * 8 + (threadIdx.x >> 5);
  if (gw >= B * H * Lq) return;
  const int lane = threadIdx.x & 31;
  const int l = gw % Lq;
  const int bh = gw / Lq;
  const int h = bh % H;
  const int b = bh / H;
  const float* qr = q + (((size_t)b * Lq + l) * H + h) * HD;
  float mx = NEG_BIG, sm = 0.f;
  for (int s = lane; s < U; s += 32) {
    const int ki = (int)(mixhash((uint32_t)l * 2654435761u + (uint32_t)s * 40503u + seed) % (uint32_t)Lk);
    const float* kr = k + (((size_t)b * Lk + ki) * H + h) * HD;
    float dot = 0.f;
#pragma unroll
    for (int d = 0; d < HD; ++d) dot += qr[d] * kr[d];
    mx = fmaxf(mx, dot);
    sm += dot;
  }
#pragma unroll
  for (int o = 16; o > 0; o >>= 1) {
    mx = fmaxf(mx, __shfl_xor(mx, o, 32));
    sm += __shfl_xor(sm, o, 32);
  }
  if (lane == 0) Mm[gw] = mx - sm / (float)Lk;
}

// Iterative block top-u (block per (b,h)).
__global__ __launch_bounds__(256)
void topk_kernel(const float* __restrict__ Mm, int* __restrict__ top, int Lq, int u) {
  extern __shared__ unsigned char smraw[];
  float* vals = (float*)smraw;                 // Lq
  float* rv = vals + Lq;                       // 256
  int* ri = (int*)(rv + 256);                  // 256
  const float* row = Mm + (size_t)blockIdx.x * Lq;
  for (int i = threadIdx.x; i < Lq; i += 256) vals[i] = row[i];
  __syncthreads();
  for (int it = 0; it < u; ++it) {
    float bv = NEG_BIG; int bi = 0;
    for (int i = threadIdx.x; i < Lq; i += 256)
      if (vals[i] > bv) { bv = vals[i]; bi = i; }
    rv[threadIdx.x] = bv; ri[threadIdx.x] = bi;
    __syncthreads();
    for (int o = 128; o > 0; o >>= 1) {
      if (threadIdx.x < o && rv[threadIdx.x + o] > rv[threadIdx.x]) {
        rv[threadIdx.x] = rv[threadIdx.x + o];
        ri[threadIdx.x] = ri[threadIdx.x + o];
      }
      __syncthreads();
    }
    if (threadIdx.x == 0) {
      top[(size_t)blockIdx.x * u + it] = ri[0];
      vals[ri[0]] = NEG_BIG;
    }
    __syncthreads();
  }
}

// Encoder context init: ctx[b,h,l,:] = mean_l(v). Block per (b,h).
__global__ __launch_bounds__(256)
void ctx_mean_kernel(const float* __restrict__ v, float* __restrict__ ctx,
                     int B, int H, int L) {
  const int h = blockIdx.x % H;
  const int b = blockIdx.x / H;
  __shared__ float part[256];
  __shared__ float mean[HD];
  const int d = threadIdx.x % HD;
  const int grp = threadIdx.x / HD;   // 4 groups
  float s = 0.f;
  for (int l = grp; l < L; l += 4) s += v[(((size_t)b * L + l) * H + h) * HD + d];
  part[threadIdx.x] = s;
  __syncthreads();
  if (threadIdx.x < HD)
    mean[threadIdx.x] = (part[threadIdx.x] + part[threadIdx.x + 64] +
                         part[threadIdx.x + 128] + part[threadIdx.x + 192]) / (float)L;
  __syncthreads();
  float* cb = ctx + (size_t)(b * H + h) * L * HD;
  for (size_t i = threadIdx.x; i < (size_t)L * HD; i += 256) cb[i] = mean[i & (HD - 1)];
}

// Decoder context init: ctx[b,h,l,:] = cumsum_l(v). Thread per (b,h,d).
__global__ void ctx_cumsum_kernel(const float* __restrict__ v, float* __restrict__ ctx,
                                  int B, int H, int L) {
  const int idx = blockIdx.x * blockDim.x + threadIdx.x;
  if (idx >= B * H * HD) return;
  const int d = idx % HD;
  const int bh = idx / HD;
  const int h = bh % H;
  const int b = bh / H;
  float run = 0.f;
  for (int l = 0; l < L; ++l) {
    run += v[(((size_t)b * L + l) * H + h) * HD + d];
    ctx[(((size_t)b * H + h) * L + l) * HD + d] = run;
  }
}

// ---------------------------------------------------------------------------
// Softmax attention row. top!=null: only selected query rows (prob-sparse
// update, optional causal mask), top==null: every query (full cross attn).
// Writes ctx[b,h,t,:]. Shared: scores[Lk] + q[64] + red[256].
// ---------------------------------------------------------------------------
__global__ __launch_bounds__(256)
void attn_row_kernel(const float* __restrict__ q, const float* __restrict__ k,
                     const float* __restrict__ v, const int* __restrict__ top,
                     float* __restrict__ ctx, int B, int H, int Lq, int Lk,
                     int u, int causal) {
  const int ui = blockIdx.x % u;
  const int bh = blockIdx.x / u;
  const int h = bh % H;
  const int b = bh / H;
  const int t = top ? top[(size_t)bh * u + ui] : ui;

  extern __shared__ unsigned char smraw[];
  float* sc = (float*)smraw;      // Lk
  float* qd = sc + Lk;            // 64
  float* red = qd + HD;           // 256

  for (int d = threadIdx.x; d < HD; d += 256)
    qd[d] = q[(((size_t)b * Lq + t) * H + h) * HD + d];
  __syncthreads();

  float lmax = NEG_BIG;
  for (int s = threadIdx.x; s < Lk; s += 256) {
    const float* kr = k + (((size_t)b * Lk + s) * H + h) * HD;
    float dot = 0.f;
#pragma unroll
    for (int d = 0; d < HD; ++d) dot += qd[d] * kr[d];
    dot *= 0.125f;                      // 1/sqrt(64)
    if (causal && s > t) dot = NEG_BIG;
    sc[s] = dot;
    lmax = fmaxf(lmax, dot);
  }
  red[threadIdx.x] = lmax;
  __syncthreads();
  for (int o = 128; o > 0; o >>= 1) {
    if (threadIdx.x < o) red[threadIdx.x] = fmaxf(red[threadIdx.x], red[threadIdx.x + o]);
    __syncthreads();
  }
  const float m = red[0];
  __syncthreads();

  float lsum = 0.f;
  for (int s = threadIdx.x; s < Lk; s += 256) {
    const float e = __expf(sc[s] - m);
    sc[s] = e;
    lsum += e;
  }
  red[threadIdx.x] = lsum;
  __syncthreads();
  for (int o = 128; o > 0; o >>= 1) {
    if (threadIdx.x < o) red[threadIdx.x] += red[threadIdx.x + o];
    __syncthreads();
  }
  const float inv = 1.0f / red[0];
  __syncthreads();

  if (threadIdx.x < HD) {
    const int d = threadIdx.x;
    float o = 0.f;
    for (int s = 0; s < Lk; ++s)
      o += sc[s] * v[(((size_t)b * Lk + s) * H + h) * HD + d];
    ctx[(((size_t)b * H + h) * Lq + t) * HD + d] = o * inv;
  }
}

// ctx (B,H,L,HD) -> out (B,L,H*HD)  (the non-mix reshape)
__global__ void ctx_to_blhd_kernel(const float* __restrict__ ctx, float* __restrict__ out,
                                   int B, int H, int L) {
  const size_t i = (size_t)blockIdx.x * blockDim.x + threadIdx.x;
  const size_t total = (size_t)B * H * L * HD;
  if (i >= total) return;
  const int d = (int)(i % HD);
  size_t r = i / HD;
  const int l = (int)(r % L); r /= L;
  const int h = (int)(r % H);
  const int b = (int)(r / H);
  out[(((size_t)b * L + l) * H + h) * HD + d] = ctx[i];
}

// ---------------------------------------------------------------------------
// Distil conv path: circular im2col (K = 512*3), then WMMA GEMM, then
// BN stats (deterministic block-serial reduction) + ELU + maxpool stride 2.
// ---------------------------------------------------------------------------
__global__ void im2col_circ_kernel(const float* __restrict__ x, float* __restrict__ a,
                                   int B, int L) {
  const size_t i = (size_t)blockIdx.x * blockDim.x + threadIdx.x;
  const size_t total = (size_t)B * L * DM * 3;
  if (i >= total) return;
  const int k = (int)(i % 3);
  size_t r = i / 3;
  const int c = (int)(r % DM); r /= DM;
  const int l = (int)(r % L);
  const int b = (int)(r / L);
  int src = l + k - 1;
  src = (src + L) % L;
  a[(((size_t)b * L + l) * DM + c) * 3 + k] = x[((size_t)b * L + src) * DM + c];
}

__global__ __launch_bounds__(256)
void bn_stats_kernel(const float* __restrict__ h, float* __restrict__ mean,
                     float* __restrict__ var, int rows) {
  const int c = blockIdx.x;
  __shared__ float s1[256], s2[256];
  float a = 0.f, b = 0.f;
  for (int r = threadIdx.x; r < rows; r += 256) {
    const float v = h[(size_t)r * DM + c];
    a += v; b += v * v;
  }
  s1[threadIdx.x] = a; s2[threadIdx.x] = b;
  __syncthreads();
  for (int o = 128; o > 0; o >>= 1) {
    if (threadIdx.x < o) { s1[threadIdx.x] += s1[threadIdx.x + o]; s2[threadIdx.x] += s2[threadIdx.x + o]; }
    __syncthreads();
  }
  if (threadIdx.x == 0) {
    const float m = s1[0] / (float)rows;
    mean[c] = m;
    var[c] = s2[0] / (float)rows - m * m;
  }
}

__global__ void bn_elu_pool_kernel(const float* __restrict__ h, const float* __restrict__ mean,
                                   const float* __restrict__ var, const float* __restrict__ g,
                                   const float* __restrict__ bb, float* __restrict__ out,
                                   int B, int L) {
  const int L2 = (L - 1) / 2 + 1;
  const size_t i = (size_t)blockIdx.x * blockDim.x + threadIdx.x;
  const size_t total = (size_t)B * L2 * DM;
  if (i >= total) return;
  const int c = (int)(i % DM);
  size_t r = i / DM;
  const int l2 = (int)(r % L2);
  const int b = (int)(r / L2);
  const float mu = mean[c];
  const float inv = rsqrtf(var[c] + 1e-5f);
  const float gg = g[c], bc = bb[c];
  float best = NEG_BIG;
#pragma unroll
  for (int k = 0; k < 3; ++k) {
    const int p = 2 * l2 + k - 1;
    if (p >= 0 && p < L) {
      float x = h[((size_t)b * L + p) * DM + c];
      x = (x - mu) * inv * gg + bc;
      x = (x > 0.f) ? x : expm1f(x);   // ELU
      best = fmaxf(best, x);
    }
  }
  out[((size_t)b * L2 + l2) * DM + c] = best;
}

// Final projection, only the last PRED rows: out (B, PRED, 7).
__global__ void proj_kernel(const float* __restrict__ y, const float* __restrict__ W,
                            const float* __restrict__ bias, float* __restrict__ out,
                            int B, int Ld) {
  const int i = blockIdx.x * blockDim.x + threadIdx.x;
  const int total = B * PRED * 7;
  if (i >= total) return;
  const int o = i % 7;
  int r = i / 7;
  const int l = r % PRED;
  const int b = r / PRED;
  const float* yr = y + ((size_t)b * Ld + (Ld - PRED) + l) * DM;
  const float* wr = W + (size_t)o * DM;
  float s = bias[o];
  for (int d = 0; d < DM; ++d) s += yr[d] * wr[d];
  out[((size_t)b * PRED + l) * 7 + o] = s;
}

// ---------------------------------------------------------------------------
// Host-side orchestration
// ---------------------------------------------------------------------------
enum {  // JAX-pytree (sorted-key) flatten order of setup_inputs()
  IN_XENC = 0, IN_MARKENC, IN_XDEC, IN_MARKDEC,
  P_DEC_C1B, P_DEC_C1W, P_DEC_C2B, P_DEC_C2W,
  P_DEC_CWK, P_DEC_CWO, P_DEC_CWQ, P_DEC_CWV,
  P_DEC_CBK, P_DEC_CBO, P_DEC_CBQ, P_DEC_CBV,
  P_DEC_LN1B, P_DEC_LN1G, P_DEC_LN2B, P_DEC_LN2G, P_DEC_LN3B, P_DEC_LN3G,
  P_DEC_SWK, P_DEC_SWO, P_DEC_SWQ, P_DEC_SWV,
  P_DEC_SBK, P_DEC_SBO, P_DEC_SBQ, P_DEC_SBV,
  P_DEC_NORM_B, P_DEC_NORM_G, P_DEC_TOK_W,
  P_DIS_B, P_DIS_BNB, P_DIS_BNG, P_DIS_W,
  P_ENC_C1B, P_ENC_C1W, P_ENC_C2B, P_ENC_C2W,
  P_ENC_WK, P_ENC_WO, P_ENC_WQ, P_ENC_WV,
  P_ENC_BK, P_ENC_BO, P_ENC_BQ, P_ENC_BV,
  P_ENC_LN1B, P_ENC_LN1G, P_ENC_LN2B, P_ENC_LN2G,
  P_ENC_NORM_B, P_ENC_NORM_G, P_ENC_TOK_W,
  P_PROJ_B, P_PROJ_W
};

static inline void gemm(hipStream_t st, const float* X, const float* W, const float* bias,
                        float* Y, int M, int N, int K, int act) {
  const size_t shbytes = (size_t)16 * K * sizeof(_Float16);  // A stripe in LDS
  gemm_wmma_kernel<<<M / 16, 256, shbytes, st>>>(X, W, bias, Y, M, N, K, act);
}
static inline void layernorm(hipStream_t st, const float* a, const float* resid,
                             const float* g, const float* b, float* out, int rows) {
  layernorm_kernel<<<(rows + 7) / 8, 256, 0, st>>>(a, resid, g, b, out, rows);
}
static inline int n_top(int L) {
  int u = 5 * (int)ceilf(logf((float)L));
  return (u > L) ? L : u;
}

extern "C" void kernel_launch(void* const* d_in, const int* in_sizes, int n_in,
                              void* d_out, int out_size, void* d_ws, size_t ws_size,
                              hipStream_t stream) {
  (void)in_sizes; (void)n_in; (void)out_size; (void)ws_size;
  const float* P[64];
  for (int i = 0; i < 58; ++i) P[i] = (const float*)d_in[i];
  const int* mark_enc = (const int*)d_in[IN_MARKENC];
  const int* mark_dec = (const int*)d_in[IN_MARKDEC];

  // Workspace arena: 7 slots of B*LENC*DM floats (16 MB each) + smalls.
  const size_t SLOT = (size_t)BATCH * LENC * DM;   // 4,194,304 floats
  float* ws = (float*)d_ws;
  float* S0 = ws;             float* S1 = ws + 1 * SLOT; float* S2 = ws + 2 * SLOT;
  float* S3 = ws + 3 * SLOT;  float* S4 = ws + 4 * SLOT; float* S5 = ws + 5 * SLOT;
  float* S6 = ws + 6 * SLOT;
  float* Mm   = ws + 7 * SLOT;                     // B*H*Lmax = 65536
  int*   top  = (int*)(Mm + 65536);                // B*H*40
  float* bnm  = (float*)(top + 4096);              // 512
  float* bnv  = bnm + 512;                         // 512

  const int B = BATCH, H = NH;

  // ------------------ Encoder ------------------
  embed_kernel<<<(int)((SLOT + 255) / 256), 256, 0, stream>>>(
      (const float*)d_in[IN_XENC], mark_enc, P[P_ENC_TOK_W], S0, B, LENC);

  int L = LENC;
  for (int i = 0; i < 3; ++i) {
    const int M = B * L;
    const size_t wOff = (size_t)i * DM * DM, bOff = (size_t)i * DM;
    gemm(stream, S0, P[P_ENC_WQ] + wOff, P[P_ENC_BQ] + bOff, S1, M, DM, DM, 0);
    gemm(stream, S0, P[P_ENC_WK] + wOff, P[P_ENC_BK] + bOff, S2, M, DM, DM, 0);
    gemm(stream, S0, P[P_ENC_WV] + wOff, P[P_ENC_BV] + bOff, S3, M, DM, DM, 0);

    const int u = n_top(L);
    prob_meas_kernel<<<(B * H * L) / 8, 256, 0, stream>>>(S1, S2, Mm, B, H, L, L, u, 1234u + i);
    topk_kernel<<<B * H, 256, (size_t)(L + 512) * 4, stream>>>(Mm, top, L, u);
    ctx_mean_kernel<<<B * H, 256, 0, stream>>>(S3, S4, B, H, L);
    attn_row_kernel<<<B * H * u, 256, (size_t)(L + 320) * 4, stream>>>(
        S1, S2, S3, top, S4, B, H, L, L, u, 0);
    ctx_to_blhd_kernel<<<(int)(((size_t)B * H * L * HD + 255) / 256), 256, 0, stream>>>(S4, S5, B, H, L);
    gemm(stream, S5, P[P_ENC_WO] + wOff, P[P_ENC_BO] + bOff, S6, M, DM, DM, 0);
    layernorm(stream, S6, S0, P[P_ENC_LN1G] + bOff, P[P_ENC_LN1B] + bOff, S1, M);

    gemm(stream, S1, P[P_ENC_C1W] + wOff, P[P_ENC_C1B] + bOff, S2, M, DM, DM, 1 /*gelu*/);
    gemm(stream, S2, P[P_ENC_C2W] + wOff, P[P_ENC_C2B] + bOff, S3, M, DM, DM, 0);
    layernorm(stream, S3, S1, P[P_ENC_LN2G] + bOff, P[P_ENC_LN2B] + bOff, S0, M);

    if (i < 2) {  // distil conv + BN + ELU + maxpool
      im2col_circ_kernel<<<(int)(((size_t)M * DM * 3 + 255) / 256), 256, 0, stream>>>(S0, S1, B, L);
      gemm(stream, S1, P[P_DIS_W] + (size_t)i * DM * DM * 3, P[P_DIS_B] + bOff, S4, M, DM, DM * 3, 0);
      bn_stats_kernel<<<DM, 256, 0, stream>>>(S4, bnm, bnv, M);
      const int L2 = (L - 1) / 2 + 1;
      bn_elu_pool_kernel<<<(int)(((size_t)B * L2 * DM + 255) / 256), 256, 0, stream>>>(
          S4, bnm, bnv, P[P_DIS_BNG] + bOff, P[P_DIS_BNB] + bOff, S0, B, L);
      L = L2;
    }
  }
  // Encoder output norm (in place: per-row values live in registers first).
  layernorm(stream, S0, nullptr, P[P_ENC_NORM_G], P[P_ENC_NORM_B], S0, B * L);
  const int Lx = L;  // 512

  // ------------------ Decoder ------------------
  embed_kernel<<<(int)(((size_t)B * LDEC * DM + 255) / 256), 256, 0, stream>>>(
      (const float*)d_in[IN_XDEC], mark_dec, P[P_DEC_TOK_W], S1, B, LDEC);

  const int Md = B * LDEC;
  for (int i = 0; i < 2; ++i) {
    const size_t wOff = (size_t)i * DM * DM, bOff = (size_t)i * DM;
    // --- masked prob-sparse self-attention (mix=True) ---
    gemm(stream, S1, P[P_DEC_SWQ] + wOff, P[P_DEC_SBQ] + bOff, S2, Md, DM, DM, 0);
    gemm(stream, S1, P[P_DEC_SWK] + wOff, P[P_DEC_SBK] + bOff, S3, Md, DM, DM, 0);
    gemm(stream, S1, P[P_DEC_SWV] + wOff, P[P_DEC_SBV] + bOff, S4, Md, DM, DM, 0);
    const int u = n_top(LDEC);
    prob_meas_kernel<<<(B * H * LDEC) / 8, 256, 0, stream>>>(S2, S3, Mm, B, H, LDEC, LDEC, u, 7777u + i);
    topk_kernel<<<B * H, 256, (size_t)(LDEC + 512) * 4, stream>>>(Mm, top, LDEC, u);
    ctx_cumsum_kernel<<<(B * H * HD + 255) / 256, 256, 0, stream>>>(S4, S5, B, H, LDEC);
    attn_row_kernel<<<B * H * u, 256, (size_t)(LDEC + 320) * 4, stream>>>(
        S2, S3, S4, top, S5, B, H, LDEC, LDEC, u, 1 /*causal*/);
    // mix=True: (B,H,L,D) contiguous reinterpreted as (B*L, 512) is exactly
    // the reference's transpose+reshape -> feed ctx straight into Wo.
    gemm(stream, S5, P[P_DEC_SWO] + wOff, P[P_DEC_SBO] + bOff, S6, Md, DM, DM, 0);
    layernorm(stream, S6, S1, P[P_DEC_LN1G] + bOff, P[P_DEC_LN1B] + bOff, S1, Md);

    // --- full cross-attention against encoder output (S0, Lx rows/batch) ---
    gemm(stream, S1, P[P_DEC_CWQ] + wOff, P[P_DEC_CBQ] + bOff, S2, Md, DM, DM, 0);
    gemm(stream, S0, P[P_DEC_CWK] + wOff, P[P_DEC_CBK] + bOff, S3, B * Lx, DM, DM, 0);
    gemm(stream, S0, P[P_DEC_CWV] + wOff, P[P_DEC_CBV] + bOff, S4, B * Lx, DM, DM, 0);
    attn_row_kernel<<<B * H * LDEC, 256, (size_t)(Lx + 320) * 4, stream>>>(
        S2, S3, S4, nullptr, S5, B, H, LDEC, Lx, LDEC, 0);
    ctx_to_blhd_kernel<<<(int)(((size_t)B * H * LDEC * HD + 255) / 256), 256, 0, stream>>>(S5, S6, B, H, LDEC);
    gemm(stream, S6, P[P_DEC_CWO] + wOff, P[P_DEC_CBO] + bOff, S2, Md, DM, DM, 0);
    layernorm(stream, S2, S1, P[P_DEC_LN2G] + bOff, P[P_DEC_LN2B] + bOff, S1, Md);

    // --- FFN ---
    gemm(stream, S1, P[P_DEC_C1W] + wOff, P[P_DEC_C1B] + bOff, S2, Md, DM, DM, 1);
    gemm(stream, S2, P[P_DEC_C2W] + wOff, P[P_DEC_C2B] + bOff, S3, Md, DM, DM, 0);
    layernorm(stream, S3, S1, P[P_DEC_LN3G] + bOff, P[P_DEC_LN3B] + bOff, S1, Md);
  }
  layernorm(stream, S1, nullptr, P[P_DEC_NORM_G], P[P_DEC_NORM_B], S1, Md);
  proj_kernel<<<(B * PRED * 7 + 255) / 256, 256, 0, stream>>>(
      S1, P[P_PROJ_W], P[P_PROJ_B], (float*)d_out, B, LDEC);
}
